// VoxelPruningResBackBone8x_73693048865257
// MI455X (gfx1250) — compile-verified, matrix operand order checked by reading, not run on hardware
//
#include <hip/hip_runtime.h>
#include <hip/hip_bf16.h>

typedef __attribute__((ext_vector_type(16))) _Float16 v16h;
typedef __attribute__((ext_vector_type(8)))  _Float16 v8h;
typedef __attribute__((ext_vector_type(8)))  float    v8f;

// ---------------------------------------------------------------------------
// Implicit-GEMM conv3d + fused BN/ReLU/mask/residual epilogue (wave32 WMMA).
// One wave computes a 16-voxel x 16-outchannel tile with
// v_wmma_f32_16x16x32_f16 accumulated over K = KD*KH*KW*CIN in 32-chunks.
// Grid: x = M-tiles / 8 waves, y = N-tiles (all waves of a block share the
// same B-fragment stream and walk adjacent voxel tiles -> L0/L2 reuse).
//
// Activations: channel-last f16  A[voxel*CIN + c]
// Weights: pre-repacked into WMMA B-fragment order:
//    Bfrag[((ntile*nChunks + chunk)*32 + lane)*16 + elem]
// so each K-step needs just two contiguous 16B loads per lane.
//
// WMMA f16 lane layout (ISA 7.12.2, wave32), half = lane/16:
//   A: row M = lane%16; elems 0..7 -> K = kb+half*8+j ; 8..15 -> K = kb+16+half*8+(j-8)
//   B: col N = lane%16; same K mapping (baked into Bfrag by the repack kernel)
//   C/D: VGPR i, lane L -> M = i + (L>=16 ? 8:0), N = L%16
//
// K ordering: k = tap*CIN + c, tap = (kd*KH+kh)*KW+kw  (matches repack).
// KD/KH/KW are template params -> tap loops fully unroll, all tap
// decompositions and chunk offsets fold to constants.
//
// MODE 0: relu(bn(conv))*mask            -> f16 channel-last
// MODE 1: relu(bn(conv)*mask + identity) -> f16 channel-last
// MODE 2: relu(bn(conv))*mask            -> f32 NCDHW (final output)
// ---------------------------------------------------------------------------
template <int MODE, int CIN, int KD, int KH, int KW>
__global__ void __launch_bounds__(256)
conv_wmma(const _Float16* __restrict__ Ain,
          const _Float16* __restrict__ Bfrag,
          const _Float16* __restrict__ Idn,
          const float* __restrict__ bnG, const float* __restrict__ bnB,
          const float* __restrict__ bnM, const float* __restrict__ bnV,
          const float* __restrict__ maskOut,
          _Float16* __restrict__ Aout, float* __restrict__ Fout,
          int Di, int Hi, int Wi,
          int Do, int Ho, int Wo, int Cout,
          int SD, int SH, int SW, int PD, int PH, int PW)
{
    const int lane = threadIdx.x & 31;
    const int wv   = threadIdx.x >> 5;
    const int hf   = lane >> 4;
    const int lr   = lane & 15;
    const int Vout = Do * Ho * Wo;
    const int nMT  = (Vout + 15) >> 4;

    const int mt = blockIdx.x * (blockDim.x >> 5) + wv;
    if (mt >= nMT) return;                    // wave-uniform
    const int nt = blockIdx.y;
    const int tileBase = mt << 4;
    const int nBase    = nt << 4;

    // This lane's A-row output voxel (clamped for the edge tile).
    int vo  = tileBase + lr;
    int voc = vo < Vout ? vo : Vout - 1;
    const int HoWo = Ho * Wo;
    int od = voc / HoWo;  int rr = voc - od * HoWo;
    int oh = rr / Wo;     int ow = rr - oh * Wo;
    const int idb = od * SD - PD, ihb = oh * SH - PH, iwb = ow * SW - PW;

    constexpr int T_  = KD * KH * KW;
    constexpr int NCH = (T_ * CIN + 31) >> 5;
    const _Float16* bp = Bfrag + (((long long)nt * NCH) * 32 + lane) * 16;

    v8f acc = {};
    const v8h zero = {};

    auto mm = [&](v8h a0, v8h a1, v8h b0, v8h b1) {
        v16h a, b;
#pragma unroll
        for (int j = 0; j < 8; ++j) {
            a[j] = a0[j]; a[j + 8] = a1[j];
            b[j] = b0[j]; b[j + 8] = b1[j];
        }
        acc = __builtin_amdgcn_wmma_f32_16x16x32_f16(
            false, a, false, b, (short)0, acc, false, false);
    };
    auto tapOff = [&](int kd, int kh, int kw, bool& ok) -> long long {
        const int id = idb + kd, ih = ihb + kh, iw = iwb + kw;
        ok = (unsigned)id < (unsigned)Di && (unsigned)ih < (unsigned)Hi &&
             (unsigned)iw < (unsigned)Wi;
        return ok ? ((long long)(id * Hi + ih) * Wi + iw) * CIN : 0;
    };

    if constexpr (CIN >= 32) {
#pragma unroll
        for (int kd = 0; kd < KD; ++kd)
#pragma unroll
            for (int kh = 0; kh < KH; ++kh)
#pragma unroll
                for (int kw = 0; kw < KW; ++kw) {
                    bool ok;
                    const long long voff = tapOff(kd, kh, kw, ok);
                    const _Float16* ap = Ain + voff + hf * 8;
#pragma unroll
                    for (int cc = 0; cc < CIN / 32; ++cc) {
                        v8h a0 = ok ? *(const v8h*)(ap + cc * 32)      : zero;
                        v8h a1 = ok ? *(const v8h*)(ap + cc * 32 + 16) : zero;
                        v8h b0 = *(const v8h*)(bp);
                        v8h b1 = *(const v8h*)(bp + 8);
                        bp += 512;
                        mm(a0, a1, b0, b1);
                    }
                }
    } else if constexpr (CIN == 16) {
        // One 32-K chunk = two consecutive taps of 16 channels.
#pragma unroll
        for (int q = 0; q < (T_ + 1) / 2; ++q) {
            const int t0 = 2 * q, t1 = 2 * q + 1;
            const int kd0 = t0 / (KH * KW), r0 = t0 % (KH * KW);
            bool ok0;
            const long long v0 = tapOff(kd0, r0 / KW, r0 % KW, ok0);
            v8h a0 = ok0 ? *(const v8h*)(Ain + v0 + hf * 8) : zero;
            v8h a1 = zero;
            if (t1 < T_) {                      // folds at compile time
                const int kd1 = t1 / (KH * KW), r1 = t1 % (KH * KW);
                bool ok1;
                const long long v1 = tapOff(kd1, r1 / KW, r1 % KW, ok1);
                a1 = ok1 ? *(const v8h*)(Ain + v1 + hf * 8) : zero;
            }
            v8h b0 = *(const v8h*)(bp);
            v8h b1 = *(const v8h*)(bp + 8);
            bp += 512;
            mm(a0, a1, b0, b1);
        }
    } else {
        // CIN == 4: input layer, per-element gather (constants after unroll).
        constexpr int Ktot = T_ * CIN;
        constexpr int KHW  = KH * KW;
        auto gather = [&](int k) -> _Float16 {
            _Float16 r = (_Float16)0.f;
            if (k < Ktot) {
                int tap = k / CIN;  int c  = k - tap * CIN;
                int kd = tap / KHW; int t2 = tap - kd * KHW;
                int kh = t2 / KW;   int kw = t2 - kh * KW;
                int id = idb + kd, ih = ihb + kh, iw = iwb + kw;
                if ((unsigned)id < (unsigned)Di && (unsigned)ih < (unsigned)Hi &&
                    (unsigned)iw < (unsigned)Wi)
                    r = Ain[((long long)(id * Hi + ih) * Wi + iw) * CIN + c];
            }
            return r;
        };
#pragma unroll
        for (int kb = 0; kb < NCH * 32; kb += 32) {
            v8h a0 = zero, a1 = zero;
#pragma unroll
            for (int j = 0; j < 8; ++j) {
                a0[j] = gather(kb + hf * 8 + j);
                a1[j] = gather(kb + 16 + hf * 8 + j);
            }
            v8h b0 = *(const v8h*)(bp);
            v8h b1 = *(const v8h*)(bp + 8);
            bp += 512;
            mm(a0, a1, b0, b1);
        }
    }

    // Fused epilogue: BN(eval) -> relu/mask [-> +identity -> relu] -> store.
    const int   ch = nBase + lr;
    const float g  = bnG[ch], be = bnB[ch], mu = bnM[ch];
    const float iv = rsqrtf(bnV[ch] + 1e-3f);
#pragma unroll
    for (int i = 0; i < 8; ++i) {
        int v = tileBase + i + hf * 8;
        if (v >= Vout) continue;            // post-WMMA divergence is fine
        float val = (acc[i] - mu) * g * iv + be;
        float mv  = maskOut[v];
        if (MODE == 0) {
            val = fmaxf(val, 0.f) * mv;
            Aout[(long long)v * Cout + ch] = (_Float16)val;
        } else if (MODE == 1) {
            val = val * mv + (float)Idn[(long long)v * Cout + ch];
            val = fmaxf(val, 0.f);
            Aout[(long long)v * Cout + ch] = (_Float16)val;
        } else {
            val = fmaxf(val, 0.f) * mv;
            Fout[(long long)ch * Vout + v] = val;   // NCDHW, N=1
        }
    }
}

// ---- weight repack: w[O][I][T] f32 -> WMMA B-fragment f16 ------------------
// Bfrag[((ntile*nChunks + q)*32 + lane)*16 + j], zero-padded past Ktot.
__global__ void repack_wfrag_kernel(const float* __restrict__ w,
                                    _Float16* __restrict__ Bf,
                                    int O, int I, int T)
{
    const int Ktot    = I * T;
    const int nChunks = (Ktot + 31) >> 5;
    const int total   = (O >> 4) * nChunks * 512;
    int idx = blockIdx.x * blockDim.x + threadIdx.x;
    if (idx >= total) return;
    const int j   = idx & 15;
    const int L   = (idx >> 4) & 31;
    const int q   = (idx >> 9) % nChunks;
    const int ntl = (idx >> 9) / nChunks;
    const int hf = L >> 4, lr = L & 15;
    const int k = q * 32 + (j < 8 ? hf * 8 + j : 16 + hf * 8 + (j - 8));
    _Float16 v = (_Float16)0.f;
    if (k < Ktot) {
        int tap = k / I;
        int c   = k - tap * I;
        int o   = ntl * 16 + lr;
        v = (_Float16)w[((long long)o * I + c) * T + tap];
    }
    Bf[idx] = v;
}

// ---- input pack: x[1,4,D,H,W] f32 -> channel-last f16 ----------------------
__global__ void pack_x_kernel(const float* __restrict__ x,
                              _Float16* __restrict__ A, int V, int C)
{
    int idx = blockIdx.x * blockDim.x + threadIdx.x;
    if (idx >= V * C) return;
    int v = idx / C, c = idx - (idx / C) * C;
    A[idx] = (_Float16)x[(long long)c * V + v];
}

// ---- SparseConv3d active-site rule: out active iff any input in RF ---------
__global__ void down_mask_kernel(const float* __restrict__ mi,
                                 float* __restrict__ mo,
                                 int Di, int Hi, int Wi, int Do, int Ho, int Wo,
                                 int KD, int KH, int KW, int SD, int SH, int SW,
                                 int PD, int PH, int PW)
{
    int vo = blockIdx.x * blockDim.x + threadIdx.x;
    int Vout = Do * Ho * Wo;
    if (vo >= Vout) return;
    int od = vo / (Ho * Wo);  int r = vo - od * (Ho * Wo);
    int oh = r / Wo;          int ow = r - oh * Wo;
    float any = 0.f;
    for (int kd = 0; kd < KD; ++kd) {
        int id = od * SD - PD + kd;
        if ((unsigned)id >= (unsigned)Di) continue;
        for (int kh = 0; kh < KH; ++kh) {
            int ih = oh * SH - PH + kh;
            if ((unsigned)ih >= (unsigned)Hi) continue;
            for (int kw = 0; kw < KW; ++kw) {
                int iw = ow * SW - PW + kw;
                if ((unsigned)iw >= (unsigned)Wi) continue;
                if (mi[((long long)id * Hi + ih) * Wi + iw] > 0.f) any = 1.f;
            }
        }
    }
    mo[vo] = any;
}

// ---------------------------------------------------------------------------
extern "C" void kernel_launch(void* const* d_in, const int* in_sizes, int n_in,
                              void* d_out, int out_size, void* d_ws,
                              size_t ws_size, hipStream_t stream)
{
    (void)in_sizes; (void)n_in; (void)out_size; (void)ws_size;
    const float* x  = (const float*)d_in[0];
    const float* M1 = (const float*)d_in[1];

    const int D1 = 41, H1 = 200, W1 = 176; const int V1 = D1 * H1 * W1;
    const int D2 = 21, H2 = 100, W2 = 88;  const int V2 = D2 * H2 * W2;
    const int D3 = 11, H3 = 50,  W3 = 44;  const int V3 = D3 * H3 * W3;
    const int D4 = 5,  H4 = 25,  W4 = 22;  const int V4 = D4 * H4 * W4;
    const int Dq = 2,  Hq = 25,  Wq = 22;  const int Vq = Dq * Hq * Wq;

    char* ws = (char*)d_ws; size_t off = 0;
    auto carve = [&](size_t bytes) -> void* {
        void* p = ws + off; off += (bytes + 255) & ~(size_t)255; return p;
    };

    _Float16* BUF0 = (_Float16*)carve((size_t)V1 * 16 * sizeof(_Float16));
    _Float16* BUF1 = (_Float16*)carve((size_t)V1 * 16 * sizeof(_Float16));
    _Float16* BUF2 = (_Float16*)carve((size_t)V1 * 16 * sizeof(_Float16));
    float* M2 = (float*)carve((size_t)V2 * 4);
    float* M3 = (float*)carve((size_t)V3 * 4);
    float* M4 = (float*)carve((size_t)V4 * 4);
    float* MO = (float*)carve((size_t)Vq * 4);

    // d_in flat index of each conv weight (setup_inputs insertion order);
    // its BN params follow at idx+1..idx+4 (gamma,beta,mean,var).
    struct WSpec { int idx, O, I, T; };
    const WSpec WS[21] = {
        {2, 16, 4, 27},                                           // conv_input
        {7, 16, 16, 27}, {12, 16, 16, 27}, {17, 16, 16, 27}, {22, 16, 16, 27},
        {27, 32, 16, 27},                                         // s2 down
        {32, 32, 32, 27}, {37, 32, 32, 27}, {42, 32, 32, 27}, {47, 32, 32, 27},
        {52, 64, 32, 27},                                         // s3 down
        {57, 64, 64, 27}, {62, 64, 64, 27}, {67, 64, 64, 27}, {72, 64, 64, 27},
        {77, 128, 64, 27},                                        // s4 down
        {82, 128, 128, 27}, {87, 128, 128, 27}, {92, 128, 128, 27}, {97, 128, 128, 27},
        {102, 128, 128, 3},                                       // conv_out
    };
    _Float16* WB[21];
    for (int i = 0; i < 21; ++i) {
        int nChunks = (WS[i].I * WS[i].T + 31) / 32;
        int n = (WS[i].O / 16) * nChunks * 512;
        WB[i] = (_Float16*)carve((size_t)n * sizeof(_Float16));
        repack_wfrag_kernel<<<(n + 255) / 256, 256, 0, stream>>>(
            (const float*)d_in[WS[i].idx], WB[i], WS[i].O, WS[i].I, WS[i].T);
    }

    pack_x_kernel<<<(V1 * 4 + 255) / 256, 256, 0, stream>>>(x, BUF0, V1, 4);

    down_mask_kernel<<<(V2 + 255) / 256, 256, 0, stream>>>(
        M1, M2, D1, H1, W1, D2, H2, W2, 3, 3, 3, 2, 2, 2, 1, 1, 1);
    down_mask_kernel<<<(V3 + 255) / 256, 256, 0, stream>>>(
        M2, M3, D2, H2, W2, D3, H3, W3, 3, 3, 3, 2, 2, 2, 1, 1, 1);
    down_mask_kernel<<<(V4 + 255) / 256, 256, 0, stream>>>(
        M3, M4, D3, H3, W3, D4, H4, W4, 3, 3, 3, 2, 2, 2, 0, 1, 1);
    down_mask_kernel<<<(Vq + 255) / 256, 256, 0, stream>>>(
        M4, MO, D4, H4, W4, Dq, Hq, Wq, 3, 1, 1, 2, 1, 1, 0, 0, 0);

    auto conv = [&](int MODE, int Cin, bool kOut,
                    const _Float16* Ain, const _Float16* Bp,
                    const _Float16* Idn, int bnIdx, const float* mout,
                    _Float16* Aout, float* Fout,
                    int Di, int Hi, int Wi,
                    int Do, int Ho, int Wo, int Cout,
                    int SD, int SH, int SW, int PD, int PH, int PW) {
        const float* g  = (const float*)d_in[bnIdx];
        const float* b  = (const float*)d_in[bnIdx + 1];
        const float* mu = (const float*)d_in[bnIdx + 2];
        const float* vr = (const float*)d_in[bnIdx + 3];
        int Vout = Do * Ho * Wo;
        int nMT  = (Vout + 15) / 16;
        dim3 gr((unsigned)((nMT + 7) / 8), (unsigned)(Cout / 16));
        dim3 bl(256);
#define CW_ARGS Ain, Bp, Idn, g, b, mu, vr, mout, Aout, Fout, \
                Di, Hi, Wi, Do, Ho, Wo, Cout,                 \
                SD, SH, SW, PD, PH, PW
        if      (kOut)                    conv_wmma<2, 128, 3, 1, 1><<<gr, bl, 0, stream>>>(CW_ARGS);
        else if (MODE == 0 && Cin == 4)   conv_wmma<0, 4,   3, 3, 3><<<gr, bl, 0, stream>>>(CW_ARGS);
        else if (MODE == 0 && Cin == 16)  conv_wmma<0, 16,  3, 3, 3><<<gr, bl, 0, stream>>>(CW_ARGS);
        else if (MODE == 1 && Cin == 16)  conv_wmma<1, 16,  3, 3, 3><<<gr, bl, 0, stream>>>(CW_ARGS);
        else if (MODE == 0 && Cin == 32)  conv_wmma<0, 32,  3, 3, 3><<<gr, bl, 0, stream>>>(CW_ARGS);
        else if (MODE == 1 && Cin == 32)  conv_wmma<1, 32,  3, 3, 3><<<gr, bl, 0, stream>>>(CW_ARGS);
        else if (MODE == 0 && Cin == 64)  conv_wmma<0, 64,  3, 3, 3><<<gr, bl, 0, stream>>>(CW_ARGS);
        else if (MODE == 1 && Cin == 64)  conv_wmma<1, 64,  3, 3, 3><<<gr, bl, 0, stream>>>(CW_ARGS);
        else if (MODE == 0 && Cin == 128) conv_wmma<0, 128, 3, 3, 3><<<gr, bl, 0, stream>>>(CW_ARGS);
        else if (MODE == 1 && Cin == 128) conv_wmma<1, 128, 3, 3, 3><<<gr, bl, 0, stream>>>(CW_ARGS);
#undef CW_ARGS
    };

    // conv_input: 4->16, subm k3 p1
    conv(0, 4,   false, BUF0, WB[0], nullptr, 3, M1, BUF1, nullptr,
         D1, H1, W1, D1, H1, W1, 16, 1, 1, 1, 1, 1, 1);
    // stage1 blocks (16ch)
    conv(0, 16,  false, BUF1, WB[1], nullptr, 8,  M1, BUF2, nullptr,
         D1, H1, W1, D1, H1, W1, 16, 1, 1, 1, 1, 1, 1);
    conv(1, 16,  false, BUF2, WB[2], BUF1,   13, M1, BUF0, nullptr,
         D1, H1, W1, D1, H1, W1, 16, 1, 1, 1, 1, 1, 1);
    conv(0, 16,  false, BUF0, WB[3], nullptr, 18, M1, BUF2, nullptr,
         D1, H1, W1, D1, H1, W1, 16, 1, 1, 1, 1, 1, 1);
    conv(1, 16,  false, BUF2, WB[4], BUF0,   23, M1, BUF1, nullptr,
         D1, H1, W1, D1, H1, W1, 16, 1, 1, 1, 1, 1, 1);
    // stage2: down 16->32 s2 p1 + blocks
    conv(0, 16,  false, BUF1, WB[5], nullptr, 28, M2, BUF0, nullptr,
         D1, H1, W1, D2, H2, W2, 32, 2, 2, 2, 1, 1, 1);
    conv(0, 32,  false, BUF0, WB[6], nullptr, 33, M2, BUF2, nullptr,
         D2, H2, W2, D2, H2, W2, 32, 1, 1, 1, 1, 1, 1);
    conv(1, 32,  false, BUF2, WB[7], BUF0,   38, M2, BUF1, nullptr,
         D2, H2, W2, D2, H2, W2, 32, 1, 1, 1, 1, 1, 1);
    conv(0, 32,  false, BUF1, WB[8], nullptr, 43, M2, BUF2, nullptr,
         D2, H2, W2, D2, H2, W2, 32, 1, 1, 1, 1, 1, 1);
    conv(1, 32,  false, BUF2, WB[9], BUF1,   48, M2, BUF0, nullptr,
         D2, H2, W2, D2, H2, W2, 32, 1, 1, 1, 1, 1, 1);
    // stage3: down 32->64 s2 p1 + blocks
    conv(0, 32,  false, BUF0, WB[10], nullptr, 53, M3, BUF1, nullptr,
         D2, H2, W2, D3, H3, W3, 64, 2, 2, 2, 1, 1, 1);
    conv(0, 64,  false, BUF1, WB[11], nullptr, 58, M3, BUF2, nullptr,
         D3, H3, W3, D3, H3, W3, 64, 1, 1, 1, 1, 1, 1);
    conv(1, 64,  false, BUF2, WB[12], BUF1,   63, M3, BUF0, nullptr,
         D3, H3, W3, D3, H3, W3, 64, 1, 1, 1, 1, 1, 1);
    conv(0, 64,  false, BUF0, WB[13], nullptr, 68, M3, BUF2, nullptr,
         D3, H3, W3, D3, H3, W3, 64, 1, 1, 1, 1, 1, 1);
    conv(1, 64,  false, BUF2, WB[14], BUF0,   73, M3, BUF1, nullptr,
         D3, H3, W3, D3, H3, W3, 64, 1, 1, 1, 1, 1, 1);
    // stage4: down 64->128 s2 pad(0,1,1) + blocks
    conv(0, 64,  false, BUF1, WB[15], nullptr, 78, M4, BUF0, nullptr,
         D3, H3, W3, D4, H4, W4, 128, 2, 2, 2, 0, 1, 1);
    conv(0, 128, false, BUF0, WB[16], nullptr, 83, M4, BUF2, nullptr,
         D4, H4, W4, D4, H4, W4, 128, 1, 1, 1, 1, 1, 1);
    conv(1, 128, false, BUF2, WB[17], BUF0,   88, M4, BUF1, nullptr,
         D4, H4, W4, D4, H4, W4, 128, 1, 1, 1, 1, 1, 1);
    conv(0, 128, false, BUF1, WB[18], nullptr, 93, M4, BUF2, nullptr,
         D4, H4, W4, D4, H4, W4, 128, 1, 1, 1, 1, 1, 1);
    conv(1, 128, false, BUF2, WB[19], BUF1,   98, M4, BUF0, nullptr,
         D4, H4, W4, D4, H4, W4, 128, 1, 1, 1, 1, 1, 1);
    // conv_out: 128->128, k(3,1,1) s(2,1,1) p0, f32 NCDHW to d_out
    conv(2, 128, true,  BUF0, WB[20], nullptr, 103, MO, nullptr, (float*)d_out,
         D4, H4, W4, Dq, Hq, Wq, 128, 2, 1, 1, 0, 0, 0);
}
